// MatrixMultApproximator_55954833932752
// MI455X (gfx1250) — compile-verified, matrix operand order checked by reading, not run on hardware
//
#include <hip/hip_runtime.h>
#include <hip/hip_bf16.h>
#include <math.h>

// ---------------------------------------------------------------------------
// Problem: OUT = IN = B = 4096.
//   T  = (V - cos(theta)*V_p)/sin(theta)
//   Tq = clip(round(T * 1/mean|T|), -1, 1)           (forward ternary values)
//   out = cos(theta) * (X @ V_p^T) + sign(X @ Tq^T) * bias
// Strategy: f32 prep (reduction + quant + bf16 convert into d_ws), then a
// single tiled bf16 WMMA GEMM computing both accumulators with shared X tiles.
// ---------------------------------------------------------------------------

#define NDIM 4096
#define NTOT (NDIM * NDIM)

typedef __attribute__((ext_vector_type(16))) __bf16 v16bf;
typedef __attribute__((ext_vector_type(8)))  float  v8f;

// ---------------- Kernel 1: zero the reduction accumulator ----------------
__global__ void zero_sum_kernel(double* s) {
    if (blockIdx.x == 0 && threadIdx.x == 0) s[0] = 0.0;
}

// ---------------- Kernel 2: sum |T| over the whole tensor ------------------
__global__ __launch_bounds__(256)
void absmean_kernel(const float* __restrict__ V, const float* __restrict__ Vp,
                    const float* __restrict__ theta, double* __restrict__ sum) {
    __shared__ double red[256];
    const float th    = theta[0];
    const float c     = __cosf(th);            // hw v_cos_f32 (theta ~ 0.2)
    const float inv_s = 1.0f / __sinf(th);

    double local = 0.0;
    const int stride = gridDim.x * blockDim.x;
    for (int i = blockIdx.x * blockDim.x + threadIdx.x; i < NTOT; i += stride) {
        float t = (V[i] - c * Vp[i]) * inv_s;
        local += (double)fabsf(t);
    }
    red[threadIdx.x] = local;
    __syncthreads();
    for (int o = 128; o > 0; o >>= 1) {
        if (threadIdx.x < o) red[threadIdx.x] += red[threadIdx.x + o];
        __syncthreads();
    }
    if (threadIdx.x == 0) atomicAdd(sum, red[0]);
}

// -------- Kernel 3: ternary-quantize T and convert operands to bf16 --------
__global__ __launch_bounds__(256)
void quant_convert_kernel(const float* __restrict__ V, const float* __restrict__ Vp,
                          const float* __restrict__ X, const float* __restrict__ theta,
                          const double* __restrict__ sum,
                          __bf16* __restrict__ Xb, __bf16* __restrict__ Vpb,
                          __bf16* __restrict__ Tqb) {
    const float th    = theta[0];
    const float c     = __cosf(th);
    const float inv_s = 1.0f / __sinf(th);
    float mean = (float)(sum[0] / (double)NTOT);
    mean = fmaxf(mean, 1e-5f);
    const float scale = 1.0f / mean;

    const int stride = gridDim.x * blockDim.x;
    for (int i = blockIdx.x * blockDim.x + threadIdx.x; i < NTOT; i += stride) {
        float vp = Vp[i];
        float t  = (V[i] - c * vp) * inv_s;
        float q  = rintf(t * scale);               // round-to-nearest-even
        q = fminf(fmaxf(q, -1.0f), 1.0f);
        Tqb[i] = (__bf16)q;
        Vpb[i] = (__bf16)vp;
        Xb[i]  = (__bf16)X[i];
    }
}

// ---------------- Kernel 4: dual-accumulator bf16 WMMA GEMM ----------------
// Block tile: M=64 (X rows) x N=256 (OUT cols), 8 waves laid out 2(M) x 4(N),
// each wave owns a 32x64 output tile = 2x4 WMMA subtiles x 2 accumulators
// (16 WMMAs per K-step per wave, A-fragments reused 8x).
#define BM 64
#define BN 256
#define BK 32
#define LDP 40   // padded LDS row stride in bf16 elements (80B, 16B-aligned)

__device__ inline v16bf load_frag(const __bf16* p0, const __bf16* p1) {
    v16bf f;
    reinterpret_cast<float4*>(&f)[0] = *reinterpret_cast<const float4*>(p0);
    reinterpret_cast<float4*>(&f)[1] = *reinterpret_cast<const float4*>(p1);
    return f;
}

__global__ __launch_bounds__(256)
void gemm_dual_kernel(const __bf16* __restrict__ Xb, const __bf16* __restrict__ Vpb,
                      const __bf16* __restrict__ Tqb, const float* __restrict__ theta,
                      const float* __restrict__ bias, float* __restrict__ out) {
    __shared__ __bf16 sX [BM * LDP];   //  5 KB
    __shared__ __bf16 sVp[BN * LDP];   // 20 KB
    __shared__ __bf16 sTq[BN * LDP];   // 20 KB

    const int t     = threadIdx.x;
    const int lane  = t & 31;
    const int wave  = t >> 5;       // 0..7
    const int waveM = wave & 1;     // 2 waves along M
    const int waveN = wave >> 1;    // 4 waves along N (64 cols each)
    const int m0 = blockIdx.x * BM;
    const int n0 = blockIdx.y * BN;

    // ISA 16-bit A/B fragment layout: row = lane%16, lane[4] picks K-half;
    // fragment = 16B chunk at K=khi*8 then 16B chunk at K=16+khi*8.
    const int r   = lane & 15;
    const int khi = lane >> 4;

    v8f acc[2][4][2] = {};          // [m-subtile][n-subtile][0=Vp,1=Tq]

    // Staging assignment: each thread moves one 16B chunk per 64-row group.
    const int ldRow = t >> 2;       // 0..63
    const int ldK   = (t & 3) * 8;  // 0,8,16,24

    for (int kb = 0; kb < NDIM; kb += BK) {
        __syncthreads();
        // Stage X tile (64x32)
        *reinterpret_cast<uint4*>(sX + ldRow * LDP + ldK) =
            *reinterpret_cast<const uint4*>(Xb + (size_t)(m0 + ldRow) * NDIM + kb + ldK);
        // Stage Vp / Tq tiles (256x32): four row-groups of 64
#pragma unroll
        for (int rr = 0; rr < 4; ++rr) {
            const int row = ldRow + rr * 64;
            *reinterpret_cast<uint4*>(sVp + row * LDP + ldK) =
                *reinterpret_cast<const uint4*>(Vpb + (size_t)(n0 + row) * NDIM + kb + ldK);
            *reinterpret_cast<uint4*>(sTq + row * LDP + ldK) =
                *reinterpret_cast<const uint4*>(Tqb + (size_t)(n0 + row) * NDIM + kb + ldK);
        }
        __syncthreads();

        // Prefetch next K tile while WMMAs run (global_prefetch_b8, no counter).
        const int kn = kb + BK;
        if (kn < NDIM) {
            __builtin_prefetch(Xb + (size_t)(m0 + ldRow) * NDIM + kn + ldK, 0, 3);
            __builtin_prefetch(Vpb + (size_t)(n0 + ldRow) * NDIM + kn + ldK, 0, 3);
            __builtin_prefetch(Tqb + (size_t)(n0 + ldRow + 128) * NDIM + kn + ldK, 0, 3);
        }

        // A fragments (reused across both B matrices and all 4 N-subtiles)
        v16bf a[2];
#pragma unroll
        for (int sm = 0; sm < 2; ++sm) {
            const __bf16* base = sX + (waveM * 32 + sm * 16 + r) * LDP;
            a[sm] = load_frag(base + khi * 8, base + 16 + khi * 8);
        }
#pragma unroll
        for (int sn = 0; sn < 4; ++sn) {
            const __bf16* bvB = sVp + (waveN * 64 + sn * 16 + r) * LDP;
            const __bf16* btB = sTq + (waveN * 64 + sn * 16 + r) * LDP;
            v16bf bvp = load_frag(bvB + khi * 8, bvB + 16 + khi * 8);
            v16bf btq = load_frag(btB + khi * 8, btB + 16 + khi * 8);
#pragma unroll
            for (int sm = 0; sm < 2; ++sm) {
                acc[sm][sn][0] = __builtin_amdgcn_wmma_f32_16x16x32_bf16(
                    false, a[sm], false, bvp, (short)0, acc[sm][sn][0], false, false);
                acc[sm][sn][1] = __builtin_amdgcn_wmma_f32_16x16x32_bf16(
                    false, a[sm], false, btq, (short)0, acc[sm][sn][1], false, false);
            }
        }
    }

    // Epilogue: out = cos(theta)*accVp + sign(accTq)*bias
    const float c = __cosf(theta[0]);
#pragma unroll
    for (int sm = 0; sm < 2; ++sm) {
#pragma unroll
        for (int sn = 0; sn < 4; ++sn) {
            const int nloc = n0 + waveN * 64 + sn * 16 + r;
            const float bv = bias[nloc];
#pragma unroll
            for (int i = 0; i < 8; ++i) {
                // C/D layout: VGPR i -> M = i + 8*khi, N = lane%16
                const int m = m0 + waveM * 32 + sm * 16 + i + 8 * khi;
                const float xv = acc[sm][sn][0][i];
                const float xt = acc[sm][sn][1][i];
                const float sg = (xt > 0.0f) ? 1.0f : ((xt < 0.0f) ? -1.0f : 0.0f);
                out[(size_t)m * NDIM + nloc] = c * xv + sg * bv;
            }
        }
    }
}

// ---------------------------------------------------------------------------
extern "C" void kernel_launch(void* const* d_in, const int* in_sizes, int n_in,
                              void* d_out, int out_size, void* d_ws, size_t ws_size,
                              hipStream_t stream) {
    const float* V     = (const float*)d_in[0];
    const float* Vp    = (const float*)d_in[1];
    const float* X     = (const float*)d_in[2];
    const float* theta = (const float*)d_in[3];
    const float* bias  = (const float*)d_in[4];
    float* out = (float*)d_out;

    // Workspace layout: [0..8): f64 reduction scalar; then three bf16 tensors.
    char* w = (char*)d_ws;
    double* sum  = (double*)w;
    __bf16* Xb   = (__bf16*)(w + 256);
    __bf16* Vpb  = (__bf16*)(w + 256 + (size_t)NTOT * 2);
    __bf16* Tqb  = (__bf16*)(w + 256 + (size_t)NTOT * 4);

    zero_sum_kernel<<<1, 64, 0, stream>>>(sum);
    absmean_kernel<<<1024, 256, 0, stream>>>(V, Vp, theta, sum);
    quant_convert_kernel<<<4096, 256, 0, stream>>>(V, Vp, X, theta, sum, Xb, Vpb, Tqb);

    dim3 grid(NDIM / BM, NDIM / BN);   // 64 x 16 blocks
    gemm_dual_kernel<<<grid, 256, 0, stream>>>(Xb, Vpb, Tqb, theta, bias, out);

    (void)in_sizes; (void)n_in; (void)out_size; (void)ws_size;
}
